// MOEModel_71382356459745
// MI455X (gfx1250) — matrix operand order repaired and checked by reading, not compiled, not run on hardware
//
#include <hip/hip_runtime.h>
#include <math.h>

// Problem constants (match reference)
#define B_    8
#define P_    50
#define N_    128
#define M_    128
#define EMB_  128
#define H_    8
#define D_    16
#define NE_   8
#define CLIP_ 10.0f
#define NBP   (B_ * P_)      // 400 (b,p) tiles

typedef __attribute__((ext_vector_type(16))) _Float16 v16h;
typedef __attribute__((ext_vector_type(8)))  _Float16 v8h;
typedef __attribute__((ext_vector_type(8)))  float    v8f;

// f16 LDS row stride in halves (136 = 8*17: keeps 16B alignment, breaks bank stride)
#define ST  136
#define STB (ST * 2)                 // row stride in bytes
#define BUFB (128 * STB)             // bytes per 128-row f16 buffer

// LDS layout (byte offsets). Aliasing: sS over sN (dead after Q GEMM),
// sO over sWk (dead after K GEMM), We^T staging over sW/Wq (dead after Q GEMM).
// sW buffer has 144 rows: rows 0..128 = We[e]^T (all 129 outputs), rows 129..143
// zeroed so the 9th MoE tile computes exact zeros in its unused columns.
#define O_SN      (0)                        // nodes f16 / attn-weight staging sS
#define O_SR      (1 * BUFB)                 // routes f16 (cols 0..127)
#define O_SW      (2 * BUFB)                 // Wq^T f16, later We[e]^T (144 rows)
#define O_SWK     (2 * BUFB + 144 * STB)     // Wk^T f16, later out_concat sO
#define O_SWV     (O_SWK + BUFB)             // Wv^T f16
#define O_SQ      (O_SWV + BUFB)             // Q   [n][feat]
#define O_SK      (O_SQ  + BUFB)             // K   [m][feat]
#define O_SV      (O_SK  + BUFB)             // V^T [feat][m]
#define O_RLAST   (O_SV  + BUFB)             // routes col 128, f32[128]
#define O_WKLAST  (O_RLAST  + 512)           // Wk row 128, f32[128]
#define O_WVLAST  (O_WKLAST + 512)           // Wv row 128, f32[128]
#define O_GATES   (O_WVLAST + 512)           // gates f32[128][8]
#define O_SCORE   (O_GATES  + 4096)          // per-token masked score f32[128]
#define O_BDOT    (O_SCORE  + 512)           // be[e] . Wfinal, f32[8]
#define O_ZERO    (O_BDOT   + 64)            // 32 zero halves for padded frags
#define O_WGT     (O_ZERO   + 64)            // Wg^T f16, 16 rows (8 zero-padded)
#define O_IMPW    (O_WGT + 16 * STB)         // per-wave importance partials f32[8][8]
#define O_RED     (O_IMPW + 256)             // softmax max/sum
#define LDS_TOTAL (O_RED + 64)               // ~287 KB < 320 KB/WGP

__device__ __forceinline__ v8f wmma16(v16h a, v16h b, v8f c) {
  return __builtin_amdgcn_wmma_f32_16x16x32_f16(false, a, false, b, (short)0, c,
                                                false, false);
}

__device__ __forceinline__ v16h combine(v8h lo, v8h hi) {
  return __builtin_shufflevector(lo, hi, 0, 1, 2, 3, 4, 5, 6, 7, 8, 9, 10, 11,
                                 12, 13, 14, 15);
}

// A fragment 16x32 f16 from row-major [row][k] storage.
// element i -> k = (i<8 ? 8*hh+i : 16+8*hh+(i-8)) : two contiguous b128 loads.
__device__ __forceinline__ v16h frag_a(const _Float16* sm, int rowBase, int kBase,
                                       int ln, int hh) {
  const _Float16* p = sm + (rowBase + ln) * ST + kBase;
  v8h lo = *(const v8h*)(p + 8 * hh);
  v8h hi = *(const v8h*)(p + 16 + 8 * hh);
  return combine(lo, hi);
}

// B fragment 32x16 f16 from [col][k] storage (k contiguous).
// element i -> k = (i<8 ? 8*hh+i : 16+8*hh+(i-8)) : two contiguous b128 loads.
__device__ __forceinline__ v16h frag_b(const _Float16* sm, int colBase, int kBase,
                                       int ln, int hh) {
  const _Float16* p = sm + (colBase + ln) * ST + kBase;
  v8h lo = *(const v8h*)(p + 8 * hh);
  v8h hi = *(const v8h*)(p + 16 + 8 * hh);
  return combine(lo, hi);
}

__global__ __launch_bounds__(256) void moe_attn_kernel(
    const float* __restrict__ nodes, const float* __restrict__ routes,
    const float* __restrict__ ninf, const float* __restrict__ rmask,
    const float* __restrict__ Wq, const float* __restrict__ Wk,
    const float* __restrict__ Wv, const float* __restrict__ Wg,
    const float* __restrict__ We, const float* __restrict__ be,
    const float* __restrict__ Wfinal, float* __restrict__ out,
    float* __restrict__ imp_part) {
  extern __shared__ char smem[];
  _Float16* sN    = (_Float16*)(smem + O_SN);
  _Float16* sS    = (_Float16*)(smem + O_SN);    // alias
  _Float16* sR    = (_Float16*)(smem + O_SR);
  _Float16* sWt   = (_Float16*)(smem + O_SW);    // Wq^T, then We^T (144 rows)
  _Float16* sWkt  = (_Float16*)(smem + O_SWK);
  _Float16* sO    = (_Float16*)(smem + O_SWK);   // alias
  _Float16* sWvt  = (_Float16*)(smem + O_SWV);
  _Float16* sQ    = (_Float16*)(smem + O_SQ);
  _Float16* sK    = (_Float16*)(smem + O_SK);
  _Float16* sVt   = (_Float16*)(smem + O_SV);
  float* sRlast   = (float*)(smem + O_RLAST);
  float* sWkLast  = (float*)(smem + O_WKLAST);
  float* sWvLast  = (float*)(smem + O_WVLAST);
  float* sGates   = (float*)(smem + O_GATES);
  float* sScore   = (float*)(smem + O_SCORE);
  float* sBdot    = (float*)(smem + O_BDOT);
  _Float16* sZero = (_Float16*)(smem + O_ZERO);
  _Float16* sWgt  = (_Float16*)(smem + O_WGT);
  float* sImpW    = (float*)(smem + O_IMPW);
  float* sRed     = (float*)(smem + O_RED);

  const int bp   = blockIdx.x;       // (b,p) tile, 0..399
  const int tid  = threadIdx.x;
  const int lane = tid & 31;
  const int wv   = tid >> 5;         // wave id 0..7 (== head id)
  const int hh   = lane >> 4;        // lane half
  const int ln   = lane & 15;

  // hint the route_mask slab (only mid-kernel HBM stream) toward the caches
  __builtin_prefetch(rmask + (size_t)bp * N_ * M_ + tid * 64, 0, 1);

  // ---------------- Phase A0: stage + f32->f16 convert ----------------
  {
    const float* nb = nodes + (size_t)bp * N_ * EMB_;
    for (int i = tid; i < N_ * EMB_; i += 256) {
      int r = i >> 7, c = i & 127;
      sN[r * ST + c] = (_Float16)nb[i];
    }
    const float* rb = routes + (size_t)bp * M_ * (EMB_ + 1);
    for (int i = tid; i < M_ * (EMB_ + 1); i += 256) {
      int r = i / 129, c = i - r * 129;
      float v = rb[i];
      if (c < 128) sR[r * ST + c] = (_Float16)v; else sRlast[r] = v;
    }
    // weights staged TRANSPOSED: sW*t[outfeat][k]
    for (int i = tid; i < 128 * 128; i += 256) {
      int r = i >> 7, c = i & 127;   // r = k, c = outfeat
      sWt[c * ST + r] = (_Float16)Wq[i];
    }
    for (int i = tid; i < 15 * 128; i += 256) {   // zero rows 129..143 (MoE tile 8 pad)
      int r = 129 + (i >> 7), c = i & 127;
      sWt[r * ST + c] = (_Float16)0.f;
    }
    for (int i = tid; i < 129 * 128; i += 256) {
      int r = i >> 7, c = i & 127;
      if (r < 128) sWkt[c * ST + r] = (_Float16)Wk[i]; else sWkLast[c] = Wk[i];
    }
    for (int i = tid; i < 129 * 128; i += 256) {
      int r = i >> 7, c = i & 127;
      if (r < 128) sWvt[c * ST + r] = (_Float16)Wv[i]; else sWvLast[c] = Wv[i];
    }
    // Wg^T, zero-padded to 16 rows so gating is one WMMA tile
    for (int i = tid; i < 128 * 8; i += 256) {
      int k = i >> 3, e = i & 7;
      sWgt[e * ST + k] = (_Float16)Wg[i];
      sWgt[(8 + e) * ST + k] = (_Float16)0.f;
    }
    if (tid < 32) sZero[tid] = (_Float16)0.f;
    if (tid < 8) {  // bdot[e] = be[e] . Wfinal  (bias folded out of MoE loop)
      float s = 0.f;
      for (int o = 0; o < 129; ++o) s += be[tid * 129 + o] * Wfinal[o];
      sBdot[tid] = s;
    }
  }
  __syncthreads();

  // ---------------- Phase A1: Q, K, V projections (WMMA) ----------------
  {
    const int nB = 16 * wv;  // this wave's activation-row strip (n or m)
    // Q^T = Wq^T @ nodes^T : A = sWt (contig), B = sN (contig), C store packed
#pragma unroll
    for (int j = 0; j < 8; ++j) {            // output-feature tile
      v8f acc = {};
#pragma unroll
      for (int kk = 0; kk < 4; ++kk) {
        v16h a = frag_a(sWt, 16 * j, 32 * kk, ln, hh);
        v16h b = frag_b(sN, nB, 32 * kk, ln, hh);
        acc = wmma16(a, b, acc);
      }
      v8h pk;
#pragma unroll
      for (int r = 0; r < 8; ++r) pk[r] = (_Float16)acc[r];
      *(v8h*)(sQ + (nB + ln) * ST + 16 * j + 8 * hh) = pk;
    }
    // K^T = Wk^T @ routes^T (+ rank-1 for feature 128)
#pragma unroll
    for (int j = 0; j < 8; ++j) {
      v8f acc = {};
#pragma unroll
      for (int kk = 0; kk < 4; ++kk) {
        v16h a = frag_a(sWkt, 16 * j, 32 * kk, ln, hh);
        v16h b = frag_b(sR, nB, 32 * kk, ln, hh);
        acc = wmma16(a, b, acc);
      }
      float rl = sRlast[nB + ln];
      v8h pk;
#pragma unroll
      for (int r = 0; r < 8; ++r)
        pk[r] = (_Float16)(acc[r] + sWkLast[16 * j + 8 * hh + r] * rl);
      *(v8h*)(sK + (nB + ln) * ST + 16 * j + 8 * hh) = pk;
    }
    // V = routes @ Wv, stored transposed sVt[feat][m] (+ rank-1)
#pragma unroll
    for (int j = 0; j < 8; ++j) {            // feature tile
      v8f acc = {};
#pragma unroll
      for (int kk = 0; kk < 4; ++kk) {
        v16h a = frag_a(sR, nB, 32 * kk, ln, hh);
        v16h b = frag_b(sWvt, 16 * j, 32 * kk, ln, hh);
        acc = wmma16(a, b, acc);
      }
      float wl = sWvLast[16 * j + ln];
      v8h pk;
#pragma unroll
      for (int r = 0; r < 8; ++r)
        pk[r] = (_Float16)(acc[r] + sRlast[nB + 8 * hh + r] * wl);
      *(v8h*)(sVt + (16 * j + ln) * ST + nB + 8 * hh) = pk;
    }
  }
  __syncthreads();

  // ---------------- Phase B: attention, wave wv == head h ----------------
  {
    const int h = wv;
    const float* rmb = rmask + (size_t)bp * N_ * M_;
    _Float16* sSw = sS + wv * (16 * ST);   // per-wave 16x128 staging, [n][m]
#pragma unroll 1
    for (int s = 0; s < 8; ++s) {
      v8f acc[8];
#pragma unroll
      for (int t = 0; t < 8; ++t) acc[t] = (v8f){};
      // scores: K-dim = D = 16, zero-padded to one K=32 WMMA
      v16h a;
      {
        v8h lo = *(const v8h*)(sQ + (16 * s + ln) * ST + h * 16 + 8 * hh);
        v8h zz = {};
        a = combine(lo, zz);
      }
#pragma unroll
      for (int t = 0; t < 8; ++t) {
        v8h lo = *(const v8h*)(sK + (16 * t + ln) * ST + h * 16 + 8 * hh);
        v8h hi = {};
        v16h b = combine(lo, hi);
        acc[t] = wmma16(a, b, acc[t]);
      }
      // scale + route_mask + row softmax + keep mask
#pragma unroll
      for (int r = 0; r < 8; ++r) {
        int n = 16 * s + r + 8 * hh;
        float rmax = -3.4e38f;
#pragma unroll
        for (int t = 0; t < 8; ++t) {
          float v = acc[t][r] * 0.25f + rmb[n * M_ + 16 * t + ln];
          acc[t][r] = v;
          rmax = fmaxf(rmax, v);
        }
#pragma unroll
        for (int mm = 1; mm < 16; mm <<= 1) rmax = fmaxf(rmax, __shfl_xor(rmax, mm, 32));
        float rsum = 0.f;
#pragma unroll
        for (int t = 0; t < 8; ++t) {
          float e = __expf(acc[t][r] - rmax);
          acc[t][r] = e;
          rsum += e;
        }
#pragma unroll
        for (int mm = 1; mm < 16; mm <<= 1) rsum += __shfl_xor(rsum, mm, 32);
        float keep = (ninf[(size_t)bp * N_ + n] == 0.f) ? 1.f : 0.f;
        float inv = keep / rsum;
#pragma unroll
        for (int t = 0; t < 8; ++t)
          sSw[(r + 8 * hh) * ST + 16 * t + ln] = (_Float16)(acc[t][r] * inv);
      }
      asm volatile("s_wait_dscnt 0" ::: "memory");  // wave-private LDS RAW
      // out^T = V^T @ w^T : A = sVt (contig), B = sSw row-major [n][m] (contig)
      v8f oacc = {};
#pragma unroll
      for (int kk = 0; kk < 4; ++kk) {
        v16h av = frag_a(sVt, h * 16, 32 * kk, ln, hh);
        v16h b  = frag_b(sSw, 0, 32 * kk, ln, hh);
        oacc = wmma16(av, b, oacc);
      }
      v8h pk;
#pragma unroll
      for (int r = 0; r < 8; ++r) pk[r] = (_Float16)oacc[r];
      *(v8h*)(sO + (16 * s + ln) * ST + h * 16 + 8 * hh) = pk;
    }
  }
  __syncthreads();

  // -------- Phase C: gating via WMMA, lane-parallel top-2 + importance --------
  {
    v8f gl = {};
#pragma unroll
    for (int kk = 0; kk < 4; ++kk) {
      v16h a = frag_a(sO, 16 * wv, 32 * kk, ln, hh);
      v16h b = frag_b(sWgt, 0, 32 * kk, ln, hh);
      gl = wmma16(a, b, gl);
    }
    float impP = 0.f;
#pragma unroll
    for (int r = 0; r < 8; ++r) {
      float v = (ln < 8) ? gl[r] : -3.4e38f;   // lane ln == expert e
      float bv = v; int bi = ln;               // top-1 (jax tie-break: low idx)
#pragma unroll
      for (int mm = 1; mm < 16; mm <<= 1) {
        float ov = __shfl_xor(bv, mm, 32);
        int   oi = __shfl_xor(bi, mm, 32);
        if (ov > bv || (ov == bv && oi < bi)) { bv = ov; bi = oi; }
      }
      float v2 = (ln == bi) ? -3.4e38f : v;    // exclude winner
      float b2v = v2; int b2i = ln;
#pragma unroll
      for (int mm = 1; mm < 16; mm <<= 1) {
        float ov = __shfl_xor(b2v, mm, 32);
        int   oi = __shfl_xor(b2i, mm, 32);
        if (ov > b2v || (ov == b2v && oi < b2i)) { b2v = ov; b2i = oi; }
      }
      float g0 = 1.f / (1.f + __expf(b2v - bv));  // softmax over top-2
      float g1 = 1.f - g0;
      float g = (ln == bi) ? g0 : ((ln == b2i) ? g1 : 0.f);
      int tok = 16 * wv + r + 8 * hh;
      if (ln < 8) sGates[tok * 8 + ln] = g;
      impP += (ln < 8) ? g : 0.f;
    }
    impP += __shfl_xor(impP, 16, 32);
    if (lane < 8) sImpW[wv * 8 + lane] = impP;
  }
  __syncthreads();
  if (tid < 8) {  // deterministic per-block importance partial
    float s = 0.f;
#pragma unroll
    for (int w = 0; w < 8; ++w) s += sImpW[w * 8 + tid];
    imp_part[bp * 8 + tid] = s;
  }

  // ------- Phase D: dense all-expert MoE, 9 WMMA tiles (col 128 in tile 8) ----
  v8f macc[9];
#pragma unroll
  for (int t = 0; t < 9; ++t) macc[t] = (v8f){};
#pragma unroll 1
  for (int e = 0; e < NE_; ++e) {
    const float* web = We + (size_t)e * 128 * 129;
    for (int i = tid; i < 128 * 129; i += 256) {   // stage We[e]^T: sWt[o][k], o<=128
      int r = i / 129, c = i - r * 129;            // r = k, c = o
      sWt[c * ST + r] = (_Float16)web[i];
    }
    __syncthreads();
    v8f acc[9];
#pragma unroll
    for (int t = 0; t < 9; ++t) acc[t] = (v8f){};
#pragma unroll
    for (int kk = 0; kk < 4; ++kk) {
      v16h a = frag_a(sO, 16 * wv, 32 * kk, ln, hh);
#pragma unroll
      for (int t = 0; t < 9; ++t) {
        v16h b = frag_b(sWt, 16 * t, 32 * kk, ln, hh);
        acc[t] = wmma16(a, b, acc[t]);
      }
    }
#pragma unroll
    for (int r = 0; r < 8; ++r) {
      int tok = 16 * wv + r + 8 * hh;
      float g = sGates[tok * 8 + e];
#pragma unroll
      for (int t = 0; t < 9; ++t) macc[t][r] += g * acc[t][r];
    }
    __syncthreads();
  }

  // ---------------- Phase E: final score + clipped tanh + mask ----------------
  {
    float wf8 = (ln == 0) ? Wfinal[128] : 0.f;
#pragma unroll
    for (int r = 0; r < 8; ++r) {
      float sred = macc[8][r] * wf8;
#pragma unroll
      for (int t = 0; t < 8; ++t) sred += macc[t][r] * Wfinal[16 * t + ln];
#pragma unroll
      for (int mm = 1; mm < 16; mm <<= 1) sred += __shfl_xor(sred, mm, 32);
      int tok = 16 * wv + r + 8 * hh;
      float gb = 0.f;
#pragma unroll
      for (int e = 0; e < 8; ++e) gb += sGates[tok * 8 + e] * sBdot[e];
      sred += gb;
      float sc = CLIP_ * tanhf(sred) + ninf[(size_t)bp * N_ + tok];
      if (ln == 0) sScore[tok] = sc;
    }
  }
  __syncthreads();
  if (wv == 0) {  // parallel block softmax stats over 128 tokens
    float a0 = sScore[lane], a1 = sScore[lane + 32];
    float a2 = sScore[lane + 64], a3 = sScore[lane + 96];
    float mx = fmaxf(fmaxf(a0, a1), fmaxf(a2, a3));
#pragma unroll
    for (int mm = 1; mm < 32; mm <<= 1) mx = fmaxf(mx, __shfl_xor(mx, mm, 32));
    float sm = __expf(a0 - mx) + __expf(a1 - mx) + __expf(a2 - mx) + __expf(a3 - mx);
#pragma unroll
    for (int mm = 1; mm < 32; mm <<= 1) sm += __shfl_xor(sm, mm, 32);
    if (lane == 0) { sRed[0] = mx; sRed[1] = sm; }
  }
  __syncthreads();
  if (tid < 128)
    out[(size_t)bp * N_ + tid] = __expf(sScore[tid] - sRed[0]) / sRed[1];
}

// Deterministic reduction of per-block importance partials -> moe_loss
__global__ void moe_loss_kernel(const float* __restrict__ imp_part,
                                float* __restrict__ out) {
  __shared__ float simp[8];
  int e = threadIdx.x;
  if (e < 8) {
    float s = 0.f;
    for (int b = 0; b < NBP; ++b) s += imp_part[b * 8 + e];
    simp[e] = s;
  }
  __syncthreads();
  if (e == 0) {
    float mean = 0.f;
    for (int i = 0; i < 8; ++i) mean += simp[i];
    mean *= (1.f / 8.f);
    float var = 0.f;
    for (int i = 0; i < 8; ++i) { float d = simp[i] - mean; var += d * d; }
    var *= (1.f / 8.f);
    out[(size_t)NBP * N_] = var / (mean * mean + 1e-10f);
  }
}

extern "C" void kernel_launch(void* const* d_in, const int* in_sizes, int n_in,
                              void* d_out, int out_size, void* d_ws, size_t ws_size,
                              hipStream_t stream) {
  (void)in_sizes; (void)n_in; (void)out_size; (void)ws_size;
  const float* nodes  = (const float*)d_in[0];
  const float* routes = (const float*)d_in[1];
  const float* ninf   = (const float*)d_in[2];
  const float* rmask  = (const float*)d_in[3];
  const float* Wq     = (const float*)d_in[4];
  const float* Wk     = (const float*)d_in[5];
  const float* Wv     = (const float*)d_in[6];
  const float* Wg     = (const float*)d_in[7];
  const float* We     = (const float*)d_in[8];
  const float* be     = (const float*)d_in[9];
  const float* Wfin   = (const float*)d_in[10];
  float* out = (float*)d_out;
  float* imp = (float*)d_ws;  // 400*8 floats of partial importance

  (void)hipFuncSetAttribute((const void*)moe_attn_kernel,
                            hipFuncAttributeMaxDynamicSharedMemorySize, LDS_TOTAL);
  moe_attn_kernel<<<NBP, 256, LDS_TOTAL, stream>>>(
      nodes, routes, ninf, rmask, Wq, Wk, Wv, Wg, We, be, Wfin, out, imp);
  moe_loss_kernel<<<1, 64, 0, stream>>>(imp, out);
}
